// QbfNewEncoder_17119739641884
// MI455X (gfx1250) — matrix operand order repaired and checked by reading, not compiled, not run on hardware
//
#include <hip/hip_runtime.h>

typedef __bf16 bf16;
typedef __attribute__((ext_vector_type(8)))  bf16  v8bf;
typedef __attribute__((ext_vector_type(16))) bf16  v16bf;
typedef __attribute__((ext_vector_type(8)))  float v8f;
typedef __attribute__((ext_vector_type(4)))  unsigned int v4u;
typedef __attribute__((ext_vector_type(8)))  int v8i;
typedef __attribute__((ext_vector_type(4)))  int v4i;

#define BM 128
#define BN 128
#define BK 32
#define LDP 40   // padded LDS row length (bf16 elems): 32 data + 8 pad = 64B + 16B

#if defined(__has_builtin)
#if __has_builtin(__builtin_amdgcn_tensor_load_to_lds) && \
    __has_builtin(__builtin_amdgcn_s_wait_tensorcnt)
#define USE_TDM 1
#endif
#endif
#ifndef USE_TDM
#define USE_TDM 0
#endif

// ---------------------------------------------------------------------------
// Generic bf16 WMMA GEMM:  C = act( [A0|A1] @ [B0;B1] + bias )
//   A is M x K row-major (split at K1 between A0/A1), bf16
//   B is K x N row-major (split at K1 between B0/B1), bf16
//   Outputs: optional fp32 Cf (ldc), optional fp32 Cf2 (ldc2), optional bf16 Cb
// 256 threads = 8 waves (wave32); waves in 4(M) x 2(N); wave tile 32x64 ->
// 2x4 grid of v_wmma_f32_16x16x32_bf16 accumulators.
// A tile is DMA'd into LDS by the Tensor Data Mover (TENSORcnt-tracked,
// overlapped with compute on the other buffer); B tile is staged transposed
// through packed ds_store_b32.
// ---------------------------------------------------------------------------
__global__ __launch_bounds__(256) void gemm_bf16_wmma(
    const bf16* __restrict__ A0, const bf16* __restrict__ A1, int lda0, int lda1,
    const bf16* __restrict__ B0, const bf16* __restrict__ B1, int ldb0, int ldb1,
    int M, int N, int K, int K1,
    const float* __restrict__ bias, int relu,
    float* __restrict__ Cf, int ldc,
    float* __restrict__ Cf2, int ldc2,
    bf16* __restrict__ Cb, int ldcb)
{
    __shared__ bf16 sA[2][BM][LDP];
    __shared__ bf16 sB[2][BN][LDP];   // stored transposed: [n][k]

    const int tid = threadIdx.x;
    const int m0  = blockIdx.y * BM;
    const int n0  = blockIdx.x * BN;

    const int lane  = tid & 31;
    const int wid   = tid >> 5;

    // staging maps: B -> 8 n x 2 k per thread
    const int b_n8  = (tid & 15) * 8;
    const int b_kl  = (tid >> 4) * 2;

#if !USE_TDM
    // fallback A staging map: 1 row x 16 k per thread
    const int a_row = tid >> 1;
    const int a_kb  = (tid & 1) * 16;

    auto loadA8 = [&](int m_g, int k_g) -> v8bf {
        v8bf v;
        if (m_g < M && (k_g + 8) <= K1) {
            v = *(const v8bf*)(A0 + (size_t)m_g * lda0 + k_g);
        } else if (m_g < M && k_g >= K1 && (k_g + 8) <= K) {
            v = *(const v8bf*)(A1 + (size_t)m_g * lda1 + (k_g - K1));
        } else {
#pragma unroll
            for (int i = 0; i < 8; ++i) {
                int kk = k_g + i;
                bf16 e = (bf16)0.0f;
                if (m_g < M) {
                    if (kk < K1)     e = A0[(size_t)m_g * lda0 + kk];
                    else if (kk < K) e = A1[(size_t)m_g * lda1 + (kk - K1)];
                }
                v[i] = e;
            }
        }
        return v;
    };
#endif

    auto loadB8 = [&](int k_g, int n_g) -> v8bf {
        v8bf v;
        const bf16* row = nullptr;
        if (k_g < K1)     row = B0 + (size_t)k_g * ldb0;
        else if (k_g < K) row = B1 + (size_t)(k_g - K1) * ldb1;
        if (row && (n_g + 8) <= N) {
            v = *(const v8bf*)(row + n_g);
        } else {
#pragma unroll
            for (int i = 0; i < 8; ++i) {
                bf16 e = (bf16)0.0f;
                if (row && (n_g + i) < N) e = row[n_g + i];
                v[i] = e;
            }
        }
        return v;
    };

    auto stage = [&](int buf, int k0) {
#if USE_TDM
        // ---- A tile via Tensor Data Mover: 128 rows x 32 k (64B rows),
        //      LDS-padded +16B per row to match LDP, OOB rows zero-filled.
        if (wid == 0) {
            const bf16* src; int koff, Kseg, lda;
            if (k0 < K1) { src = A0; koff = k0;      Kseg = K1;     lda = lda0; }
            else         { src = A1; koff = k0 - K1; Kseg = K - K1; lda = lda1; }
            unsigned long long ga = (unsigned long long)(size_t)src +
                ((unsigned long long)(unsigned)m0 * (unsigned)lda +
                 (unsigned)koff) * 2ull;
            unsigned ldsa = (unsigned)(size_t)&sA[buf][0][0]; // low 32b = LDS offset
            unsigned td0  = (unsigned)(Kseg - koff);   // remaining K in segment
            unsigned td1  = (unsigned)(M - m0);        // remaining rows
            v4u g0;
            g0[0] = 1u;                                  // count=1, user mode
            g0[1] = ldsa;                                // lds_addr (bytes)
            g0[2] = (unsigned)(ga & 0xffffffffull);      // global_addr lo
            g0[3] = (unsigned)(ga >> 32) | (2u << 30);   // global_addr hi | type=2
            v8i g1;
            g1[0] = (int)0x06D10000u;  // data_size=2B, pad_en, 4dw pad per 16dw
            g1[1] = (int)((td0 & 0xffffu) << 16);                 // tensor_dim0 lo
            g1[2] = (int)((td0 >> 16) | ((td1 & 0xffffu) << 16)); // td0 hi | td1 lo
            g1[3] = (int)((td1 >> 16) | (32u << 16));             // td1 hi | tile_dim0=32
            g1[4] = (int)128u;                                    // tile_dim1=128
            g1[5] = (int)(unsigned)lda;                           // dim0 stride lo
            g1[6] = 0;                                            // stride hi
            g1[7] = 0;
            v4i gz4 = {0, 0, 0, 0};
            v8i gz8 = {0, 0, 0, 0, 0, 0, 0, 0};
            // 6-arg toolchain variant: (g0, g1, g2, g3, extra, cpol)
            __builtin_amdgcn_tensor_load_to_lds(g0, g1, gz4, gz4, gz8, 0);
        }
#else
#pragma unroll
        for (int j = 0; j < 2; ++j) {
            int kl = a_kb + j * 8;
            v8bf v = loadA8(m0 + a_row, k0 + kl);
            *(v8bf*)&sA[buf][a_row][kl] = v;
        }
#endif
        // ---- B tile: read rows coalesced, write transposed as packed dwords
        {
            v8bf r0 = loadB8(k0 + b_kl,     n0 + b_n8);
            v8bf r1 = loadB8(k0 + b_kl + 1, n0 + b_n8);
#pragma unroll
            for (int i = 0; i < 8; ++i) {
                unsigned short u0 = __builtin_bit_cast(unsigned short, r0[i]);
                unsigned short u1 = __builtin_bit_cast(unsigned short, r1[i]);
                unsigned int pk = (unsigned int)u0 | ((unsigned int)u1 << 16);
                *(unsigned int*)&sB[buf][b_n8 + i][b_kl] = pk;
            }
        }
    };

    const int wm0   = (wid >> 1) * 32;
    const int wn0   = (wid & 1) * 64;
    const int lrow  = lane & 15;
    const int khalf = lane >> 4;

    const v8f vzero = {0.f, 0.f, 0.f, 0.f, 0.f, 0.f, 0.f, 0.f};
    v8f acc[2][4];
#pragma unroll
    for (int i = 0; i < 2; ++i)
#pragma unroll
        for (int j = 0; j < 4; ++j) acc[i][j] = vzero;

    auto compute = [&](int buf) {
        // A frags: lanes 0-15 -> K {0..7,16..23}; lanes 16-31 -> K {8..15,24..31}
        v16bf af[2];
#pragma unroll
        for (int sm = 0; sm < 2; ++sm) {
            const bf16* p = &sA[buf][wm0 + sm * 16 + lrow][khalf * 8];
            v8bf lo = *(const v8bf*)p;
            v8bf hi = *(const v8bf*)(p + 16);
            af[sm] = __builtin_shufflevector(lo, hi,
                0,1,2,3,4,5,6,7,8,9,10,11,12,13,14,15);
        }
        // B frags from transposed LDS: lane col = lane&15, K-half by lane>>4
#pragma unroll
        for (int sn = 0; sn < 4; ++sn) {
            const bf16* p = &sB[buf][wn0 + sn * 16 + lrow][khalf * 8];
            v8bf lo = *(const v8bf*)p;
            v8bf hi = *(const v8bf*)(p + 16);
            v16bf bfrag = __builtin_shufflevector(lo, hi,
                0,1,2,3,4,5,6,7,8,9,10,11,12,13,14,15);
#pragma unroll
            for (int sm = 0; sm < 2; ++sm) {
                acc[sm][sn] = __builtin_amdgcn_wmma_f32_16x16x32_bf16(
                    false, af[sm], false, bfrag, (short)0, acc[sm][sn],
                    false, false);
            }
        }
    };

    const int numK = (K + BK - 1) / BK;
    stage(0, 0);
#if USE_TDM
    __builtin_amdgcn_s_wait_tensorcnt(0);
#endif
    __syncthreads();
    int buf = 0;
    for (int kt = 0; kt < numK; ++kt) {
        if (kt + 1 < numK) stage(buf ^ 1, (kt + 1) * BK);
        compute(buf);
#if USE_TDM
        __builtin_amdgcn_s_wait_tensorcnt(0);
#endif
        __syncthreads();
        buf ^= 1;
    }

    // Epilogue: 16x16 f32 C layout -> VGPR i: lanes0-15 M=i, lanes16-31 M=8+i
#pragma unroll
    for (int sm = 0; sm < 2; ++sm)
#pragma unroll
        for (int sn = 0; sn < 4; ++sn)
#pragma unroll
            for (int i = 0; i < 8; ++i) {
                int r = m0 + wm0 + sm * 16 + khalf * 8 + i;
                int c = n0 + wn0 + sn * 16 + lrow;
                if (r < M && c < N) {
                    float v = acc[sm][sn][i];
                    if (bias) v += bias[c];
                    if (relu) v = fmaxf(v, 0.0f);
                    if (Cf)  Cf [(size_t)r * ldc  + c] = v;
                    if (Cf2) Cf2[(size_t)r * ldc2 + c] = v;
                    if (Cb)  Cb [(size_t)r * ldcb + c] = (bf16)v;
                }
            }
}

// fp32 [R,Ccol] -> bf16 row-major + bf16 transposed, LDS-tiled for coalescing
__global__ __launch_bounds__(256) void convert_cmat_kernel(
    const float* __restrict__ in, bf16* __restrict__ out_rm,
    bf16* __restrict__ out_tr, int R, int Ccol)
{
    __shared__ float tile[32][33];
    int tx = threadIdx.x & 31;
    int ty = threadIdx.x >> 5;       // 0..7
    int r0 = blockIdx.y * 32;
    int c0 = blockIdx.x * 32;
#pragma unroll
    for (int p = 0; p < 4; ++p) {
        int r = r0 + ty + p * 8;
        int c = c0 + tx;
        float v = 0.0f;
        if (r < R && c < Ccol) v = in[(size_t)r * Ccol + c];
        tile[ty + p * 8][tx] = v;
        if (r < R && c < Ccol) out_rm[(size_t)r * Ccol + c] = (bf16)v;
    }
    __syncthreads();
#pragma unroll
    for (int p = 0; p < 4; ++p) {
        int c = c0 + ty + p * 8;     // original column -> row of out_tr
        int r = r0 + tx;
        if (c < Ccol && r < R)
            out_tr[(size_t)c * R + r] = (bf16)tile[tx][ty + p * 8];
    }
}

__global__ void cast_f32_to_bf16_strided(
    const float* __restrict__ src, int lds_,
    bf16* __restrict__ dst, int ldd, int M, int N)
{
    int c = blockIdx.x * blockDim.x + threadIdx.x;
    int r = blockIdx.y;
    if (r < M && c < N)
        dst[(size_t)r * ldd + c] = (bf16)src[(size_t)r * lds_ + c];
}

// W: [N,K] fp32 row-major  ->  WT: [K,N] bf16 row-major
__global__ void transpose_cast_w(const float* __restrict__ W,
                                 bf16* __restrict__ WT, int N, int K)
{
    int idx = blockIdx.x * blockDim.x + threadIdx.x;
    if (idx < N * K) {
        int k = idx / N;
        int n = idx - k * N;
        WT[idx] = (bf16)W[(size_t)n * K + k];
    }
}

__global__ void init_vars(const float* __restrict__ vlab,
                          float* __restrict__ pos, float* __restrict__ neg)
{
    int r = blockIdx.x;
    int c = threadIdx.x;          // 64 threads
    float v = vlab[(size_t)r * 64 + c];
    pos[(size_t)r * 832 + c] = v;
    neg[(size_t)r * 832 + c] = v;
}

extern "C" void kernel_launch(void* const* d_in, const int* in_sizes, int n_in,
                              void* d_out, int out_size, void* d_ws, size_t ws_size,
                              hipStream_t stream)
{
    const int V = 4000, C = 8000, VLAB = 64, CLAB = 64, VEMB = 128, CEMB = 256;
    const int DOUT = VLAB + 3 * 2 * VEMB;   // 832

    const float* vlabels  = (const float*)d_in[0];
    const float* clabels  = (const float*)d_in[1];
    const float* cmat_pos = (const float*)d_in[2];
    const float* cmat_neg = (const float*)d_in[3];

    float* pos = (float*)d_out;
    float* neg = pos + (size_t)V * DOUT;

    char* w = (char*)d_ws;
    size_t off = 0;
    auto alloc = [&](size_t bytes) -> void* {
        void* p = w + off;
        off += (bytes + 255) & ~(size_t)255;
        return p;
    };
    bf16* cmatP  = (bf16*)alloc((size_t)C * V * 2);
    bf16* cmatN  = (bf16*)alloc((size_t)C * V * 2);
    bf16* cmatPT = (bf16*)alloc((size_t)C * V * 2);
    bf16* cmatNT = (bf16*)alloc((size_t)C * V * 2);
    bf16* posv   = (bf16*)alloc((size_t)V * 576 * 2);
    bf16* negv   = (bf16*)alloc((size_t)V * 576 * 2);
    bf16* aggb   = (bf16*)alloc((size_t)C * 576 * 2);
    bf16* ct     = (bf16*)alloc((size_t)C * 320 * 2);
    bf16* pvb    = (bf16*)alloc((size_t)V * 320 * 2);
    bf16* nvb    = (bf16*)alloc((size_t)V * 320 * 2);
    bf16* WLT    = (bf16*)alloc((size_t)576 * 256 * 2);
    bf16* WCT    = (bf16*)alloc((size_t)320 * 128 * 2);

    {   // one-shot bf16 conversion (row-major + transposed) of both cmats
        dim3 grid(V / 32, C / 32);
        convert_cmat_kernel<<<grid, 256, 0, stream>>>(cmat_pos, cmatP, cmatPT, C, V);
        convert_cmat_kernel<<<grid, 256, 0, stream>>>(cmat_neg, cmatN, cmatNT, C, V);
    }
    init_vars<<<V, 64, 0, stream>>>(vlabels, pos, neg);
    {   // clabels -> ct[:, 0:64] (constant part of c_t)
        dim3 grid((CLAB + 255) / 256, C);
        cast_f32_to_bf16_strided<<<grid, 256, 0, stream>>>(clabels, CLAB, ct, 320, C, CLAB);
    }

    auto gemm = [&](const bf16* A0, const bf16* A1, int lda0, int lda1,
                    const bf16* B0, const bf16* B1, int ldb0, int ldb1,
                    int M, int N, int K, int K1,
                    const float* bias, int relu,
                    float* Cf, int ldc, float* Cf2, int ldc2,
                    bf16* Cb, int ldcb) {
        dim3 grid((N + BN - 1) / BN, (M + BM - 1) / BM);
        gemm_bf16_wmma<<<grid, 256, 0, stream>>>(A0, A1, lda0, lda1,
                                                 B0, B1, ldb0, ldb1,
                                                 M, N, K, K1, bias, relu,
                                                 Cf, ldc, Cf2, ldc2, Cb, ldcb);
    };

    for (int t = 0; t < 3; ++t) {
        int d = VLAB + 2 * t * VEMB;   // 64, 320, 576
        const float* W_L = (const float*)d_in[4 + 4 * t];
        const float* B_L = (const float*)d_in[5 + 4 * t];
        const float* W_C = (const float*)d_in[6 + 4 * t];
        const float* B_C = (const float*)d_in[7 + 4 * t];

        {   // current pos/neg vars -> bf16 (packed leading dim d)
            dim3 grid((d + 255) / 256, V);
            cast_f32_to_bf16_strided<<<grid, 256, 0, stream>>>(pos, DOUT, posv, d, V, d);
            cast_f32_to_bf16_strided<<<grid, 256, 0, stream>>>(neg, DOUT, negv, d, V, d);
        }
        {   // tiny weight transposes
            int nE1 = CEMB * d;
            transpose_cast_w<<<(nE1 + 255) / 256, 256, 0, stream>>>(W_L, WLT, CEMB, d);
            int nE2 = VEMB * (CLAB + CEMB);
            transpose_cast_w<<<(nE2 + 255) / 256, 256, 0, stream>>>(W_C, WCT, VEMB, CLAB + CEMB);
        }

        // agg = cmat_pos@posv + cmat_neg@negv  (fused via K-split), bf16 out
        gemm(cmatP, cmatN, V, V, posv, negv, d, d,
             C, d, 2 * V, V,
             nullptr, 0, nullptr, 0, nullptr, 0, aggb, d);

        // c_pre = relu(agg @ W_L^T + B_L) -> ct[:, 64:320] (bf16)
        gemm(aggb, aggb, d, d, WLT, WLT, CEMB, CEMB,
             C, CEMB, d, d,
             B_L, 1, nullptr, 0, nullptr, 0, ct + CLAB, CLAB + CEMB);

        // pv = cmat_pos^T @ ct ; nv = cmat_neg^T @ ct   (bf16 out)
        gemm(cmatPT, cmatPT, C, C, ct, ct, 320, 320,
             V, 320, C, C,
             nullptr, 0, nullptr, 0, nullptr, 0, pvb, 320);
        gemm(cmatNT, cmatNT, C, C, ct, ct, 320, 320,
             V, 320, C, C,
             nullptr, 0, nullptr, 0, nullptr, 0, nvb, 320);

        // pv_pre = relu(pv@W_C^T + B_C) -> pos[:, d:d+128], neg[:, d+128:d+256]
        gemm(pvb, pvb, 320, 320, WCT, WCT, VEMB, VEMB,
             V, VEMB, 320, 320,
             B_C, 1, pos + d, DOUT, neg + d + VEMB, DOUT, nullptr, 0);
        // nv_pre -> pos[:, d+128:d+256], neg[:, d:d+128]
        gemm(nvb, nvb, 320, 320, WCT, WCT, VEMB, VEMB,
             V, VEMB, 320, 320,
             B_C, 1, pos + d + VEMB, DOUT, neg + d, DOUT, nullptr, 0);
    }
    (void)in_sizes; (void)n_in; (void)out_size; (void)ws_size;
}
